// PredictModel_26963804685199
// MI455X (gfx1250) — matrix operand order repaired and checked by reading, not compiled, not run on hardware
//
#include <hip/hip_runtime.h>
#include <hip/hip_bf16.h>
#include <math.h>

#define NND 8192
#define INF_ 512
#define HF 256
#define ME 65536

typedef __attribute__((ext_vector_type(16))) __bf16 v16bf;
typedef __attribute__((ext_vector_type(8)))  float  v8f;

// packed fp32x2 -> bf16x2 (RNE) in one dword
__device__ __forceinline__ unsigned cvt2bf(float a, float b) {
#if __has_builtin(__builtin_amdgcn_cvt_pk_bf16_f32)
  typedef __attribute__((ext_vector_type(2))) __bf16 v2bf;
  v2bf r = __builtin_amdgcn_cvt_pk_bf16_f32(a, b);
  return __builtin_bit_cast(unsigned, r);
#else
  unsigned short x = __builtin_bit_cast(unsigned short, (__bf16)a);
  unsigned short y = __builtin_bit_cast(unsigned short, (__bf16)b);
  return (unsigned)x | ((unsigned)y << 16);
#endif
}

struct BF16x16 { uint4 lo; uint4 hi; };
struct U8x4    { unsigned u[8]; };

__device__ __forceinline__ unsigned lds_addr_of(const void* p) {
  // low 32 bits of a flat shared pointer == workgroup-relative LDS byte address
  return (unsigned)(unsigned long long)p;
}

// C[M x N] = A[M x K] * B[K x N], fp32 in / bf16 WMMA / fp32 out.
// Block tile: 32(M) x 256(N) x 32(K). 8 waves: 2(M) x 4(N), each 16x64.
// A tile is streamed via GLOBAL_LOAD_ASYNC_TO_LDS_B128 (ASYNCcnt), double-buffered.
// mode 1: C = elu((acc + Yadd[row,col]) / deg[row])   (fused GCN layer epilogue)
__global__ __launch_bounds__(256)
void gemm_bf16(const float* __restrict__ A, int lda,
               const float* __restrict__ B, int ldb,
               float* __restrict__ C, int ldc, int K,
               int mode, const float* __restrict__ Yadd,
               const float* __restrict__ deg)
{
  __shared__ __align__(16) float          Af[2][32 * 32];    // fp32 A tiles (async dest)
  __shared__ __align__(16) unsigned short Bs[2][256 * 32];   // bf16 B tiles, [n][k]
  const int tid   = threadIdx.x;
  const int lane  = tid & 31;
  const int wid   = tid >> 5;
  const int waveM = wid >> 2;       // 0..1 -> 16-row slice
  const int waveN = wid & 3;        // 0..3 -> 64-col slice
  const int mbase = blockIdx.y * 32;
  const int nbase = blockIdx.x * 256;
  const int lhalf = lane >> 4;      // 0/1
  const int l16   = lane & 15;

  v8f acc[4];
#pragma unroll
  for (int j = 0; j < 4; j++) {
    v8f z = {0.f,0.f,0.f,0.f,0.f,0.f,0.f,0.f};
    acc[j] = z;
  }

  // ---- staging helpers ----
  // A: 32x32 fp32 = 256 x b128; one async copy per thread, no VGPR round-trip.
  auto stageA = [&](int k0, int buf) {
    int r  = tid >> 3;
    int c4 = (tid & 7) << 2;
    const float* gp = A + (size_t)(mbase + r) * lda + k0 + c4;
    if (k0 + 64 < K) __builtin_prefetch(gp + 64, 0, 1);     // pull tile k+2 into L2
    unsigned lds = lds_addr_of(&Af[buf][r * 32 + c4]);
    asm volatile("global_load_async_to_lds_b128 %0, %1, off"
                 :: "v"(lds), "v"(gp) : "memory");
  };
  // B: 32x256 fp32 -> transposed bf16 [n][k]; pack K-pairs so stores are dword b32.
  auto stageB = [&](int k0, int buf) {
#pragma unroll
    for (int i = 0; i < 4; i++) {
      int idx = tid + i * 256;          // 1024 tasks: 16 k-pairs x 64 col-quads
      int kp = idx >> 6;                // 0..15 -> rows 2kp, 2kp+1
      int c4 = (idx & 63) << 2;
      const float* b0 = B + (size_t)(k0 + 2 * kp) * ldb + nbase + c4;
      float4 fa = *(const float4*)b0;
      float4 fb = *(const float4*)(b0 + ldb);
      *(unsigned*)&Bs[buf][(c4 + 0) * 32 + 2 * kp] = cvt2bf(fa.x, fb.x);
      *(unsigned*)&Bs[buf][(c4 + 1) * 32 + 2 * kp] = cvt2bf(fa.y, fb.y);
      *(unsigned*)&Bs[buf][(c4 + 2) * 32 + 2 * kp] = cvt2bf(fa.z, fb.z);
      *(unsigned*)&Bs[buf][(c4 + 3) * 32 + 2 * kp] = cvt2bf(fa.w, fb.w);
    }
  };

  // ---- prologue: stage tile 0 ----
  stageA(0, 0);
  stageB(0, 0);
  asm volatile("s_wait_asynccnt 0x0" ::: "memory");
  __syncthreads();

  const int nIter = K >> 5;
  for (int it = 0; it < nIter; it++) {
    const int cur = it & 1;
    const int k0  = it << 5;
    if (it + 1 < nIter) {               // kick next tile while computing this one
      stageA(k0 + 32, cur ^ 1);
      stageB(k0 + 32, cur ^ 1);
    }

    // ---- B fragments first: issue all 8 ds_load_b128 up front ----
    BF16x16 u0, u1, u2, u3;
    {
      int kofs = lhalf * 16;            // lanes>=16 hold K 16..31
      const unsigned short* bp = &Bs[cur][(waveN * 64 + l16) * 32 + kofs];
      u0.lo = *(const uint4*)(bp +  0 * 32);
      u0.hi = *(const uint4*)(bp +  0 * 32 + 8);
      u1.lo = *(const uint4*)(bp + 16 * 32);
      u1.hi = *(const uint4*)(bp + 16 * 32 + 8);
      u2.lo = *(const uint4*)(bp + 32 * 32);
      u2.hi = *(const uint4*)(bp + 32 * 32 + 8);
      u3.lo = *(const uint4*)(bp + 48 * 32);
      u3.hi = *(const uint4*)(bp + 48 * 32 + 8);
    }

    // ---- A fragment: fp32 from LDS, convert+pack to bf16 (covers B load latency) ----
    int row = waveM * 16 + l16;
    const float* ap = &Af[cur][row * 32 + lhalf * 16];  // K kofs..kofs+15
    float4 f0 = *(const float4*)(ap);
    float4 f1 = *(const float4*)(ap + 4);
    float4 f2 = *(const float4*)(ap + 8);
    float4 f3 = *(const float4*)(ap + 12);
    U8x4 t;
    t.u[0] = cvt2bf(f0.x, f0.y); t.u[1] = cvt2bf(f0.z, f0.w);
    t.u[2] = cvt2bf(f1.x, f1.y); t.u[3] = cvt2bf(f1.z, f1.w);
    t.u[4] = cvt2bf(f2.x, f2.y); t.u[5] = cvt2bf(f2.z, f2.w);
    t.u[6] = cvt2bf(f3.x, f3.y); t.u[7] = cvt2bf(f3.z, f3.w);
    v16bf afr = __builtin_bit_cast(v16bf, t);

    v16bf b0 = __builtin_bit_cast(v16bf, u0);
    v16bf b1 = __builtin_bit_cast(v16bf, u1);
    v16bf b2 = __builtin_bit_cast(v16bf, u2);
    v16bf b3 = __builtin_bit_cast(v16bf, u3);
    acc[0] = __builtin_amdgcn_wmma_f32_16x16x32_bf16(false, afr, false, b0, (short)0, acc[0], false, false);
    acc[1] = __builtin_amdgcn_wmma_f32_16x16x32_bf16(false, afr, false, b1, (short)0, acc[1], false, false);
    acc[2] = __builtin_amdgcn_wmma_f32_16x16x32_bf16(false, afr, false, b2, (short)0, acc[2], false, false);
    acc[3] = __builtin_amdgcn_wmma_f32_16x16x32_bf16(false, afr, false, b3, (short)0, acc[3], false, false);

    if (it + 1 < nIter)
      asm volatile("s_wait_asynccnt 0x0" ::: "memory");   // next A tile resident
    __syncthreads();                                      // + dscnt wait by compiler
  }

  // ---- epilogue (C layout: VGPR r -> row r + 8*(lane>=16), col = lane&15) ----
#pragma unroll
  for (int bn = 0; bn < 4; bn++)
#pragma unroll
    for (int r = 0; r < 8; r++) {
      int row = mbase + waveM * 16 + lhalf * 8 + r;
      int col = nbase + waveN * 64 + bn * 16 + l16;
      float v = acc[bn][r];
      if (mode == 1) {
        v = (v + Yadd[(size_t)row * ldc + col]) / deg[row];
        v = v > 0.f ? v : (__expf(v) - 1.f);
      }
      C[(size_t)row * ldc + col] = v;
    }
}

// deg[i] = 1 + sum_r adj[r][i]  (two-pass, deterministic)
__global__ void adj_colsum_part(const float* __restrict__ adj, float* __restrict__ part) {
  int j  = blockIdx.x * 256 + threadIdx.x;
  int r0 = blockIdx.y * 256;
  float s = 0.f;
  for (int r = 0; r < 256; r++) s += adj[(size_t)(r0 + r) * NND + j];
  part[blockIdx.y * NND + j] = s;
}
__global__ void adj_deg_fin(const float* __restrict__ part, float* __restrict__ deg) {
  int j = blockIdx.x * 256 + threadIdx.x;
  float s = 1.0f;
  for (int b = 0; b < 32; b++) s += part[b * NND + j];
  deg[j] = s;
}

// w2 = W @ a2  (512-vector)
__global__ void w_dot_a2(const float* __restrict__ W, const float* __restrict__ a2,
                         float* __restrict__ w2) {
  int r = blockIdx.x * 256 + threadIdx.x;
  float s = 0.f;
  for (int c = 0; c < HF; c++) s += W[(size_t)r * HF + c] * a2[c];
  w2[r] = s;
}
// s2[row] = x[row] . w2
__global__ void s2_kernel(const float* __restrict__ x, const float* __restrict__ w2,
                          float* __restrict__ s2) {
  __shared__ float red[256];
  int row = blockIdx.x, t = threadIdx.x;
  float p = x[(size_t)row * INF_ + t] * w2[t] + x[(size_t)row * INF_ + t + 256] * w2[t + 256];
  red[t] = p; __syncthreads();
  for (int s = 128; s > 0; s >>= 1) { if (t < s) red[t] += red[t + s]; __syncthreads(); }
  if (t == 0) s2[row] = red[0];
}

// top-16 indices of s2 -> hub list + flags (the entire sim-graph structure)
__global__ void topk_kernel(const float* __restrict__ s2, int* __restrict__ hubs,
                            int* __restrict__ hubflag) {
  __shared__ float sv[NND];
  __shared__ float rv[256];
  __shared__ int   ri[256];
  __shared__ int   win[16];
  int t = threadIdx.x;
  for (int i = t; i < NND; i += 256) sv[i] = s2[i];
  __syncthreads();
  for (int k = 0; k < 16; k++) {
    float bv = -3.4e38f; int bi = 0;
    for (int i = t; i < NND; i += 256) { float v = sv[i]; if (v > bv) { bv = v; bi = i; } }
    rv[t] = bv; ri[t] = bi; __syncthreads();
    for (int s = 128; s > 0; s >>= 1) {
      if (t < s && rv[t + s] > rv[t]) { rv[t] = rv[t + s]; ri[t] = ri[t + s]; }
      __syncthreads();
    }
    if (t == 0) { win[k] = ri[0]; sv[ri[0]] = -3.4e38f; }
    __syncthreads();
  }
  for (int i = t; i < NND; i += 256) hubflag[i] = 0;
  __syncthreads();
  if (t < 16) { hubs[t] = win[t]; hubflag[win[t]] = 1; }
}

// column partial sums of Y (8192x256)
__global__ void colsum_part(const float* __restrict__ Y, float* __restrict__ part) {
  int t = threadIdx.x;
  int r0 = blockIdx.x * 128;
  float s = 0.f;
  for (int r = 0; r < 128; r++) s += Y[(size_t)(r0 + r) * HF + t];
  part[blockIdx.x * HF + t] = s;
}
__global__ void colsum_fin(const float* __restrict__ part, const float* __restrict__ Y,
                           const int* __restrict__ hubs, float* __restrict__ TS,
                           float* __restrict__ SH) {
  int t = threadIdx.x;
  float s = 0.f;
  for (int b = 0; b < 64; b++) s += part[b * HF + t];
  TS[t] = s;
  float sh = 0.f;
  for (int j = 0; j < 16; j++) sh += Y[(size_t)hubs[j] * HF + t];
  SH[t] = sh;
}

// GCN2 layer via hub structure: hub row -> TS/8192, else (SH + Y[i])/17; then ELU
__global__ void gcn2_ew(const float* __restrict__ Y, const float* __restrict__ TS,
                        const float* __restrict__ SH, const int* __restrict__ hubflag,
                        float* __restrict__ Z) {
  int i = blockIdx.x, c = threadIdx.x;
  float v = hubflag[i] ? TS[c] * (1.f / 8192.f)
                       : (SH[c] + Y[(size_t)i * HF + c]) * (1.f / 17.f);
  v = v > 0.f ? v : (__expf(v) - 1.f);
  Z[(size_t)i * HF + c] = v;
}

// logits[m] = P[src].(h[dst] ++ h_[dst]) + b ; predict ; per-block loss partials
__global__ void edge_kernel(const int* __restrict__ src, const int* __restrict__ dst,
                            const float* __restrict__ labels, const float* __restrict__ P,
                            const float* __restrict__ H, const float* __restrict__ H2,
                            const float* __restrict__ bb, float* __restrict__ out,
                            float* __restrict__ lossp) {
  __shared__ float red[256];
  int t = threadIdx.x;
  int m = blockIdx.x * 256 + t;
  int s = src[m], d = dst[m];
  const float4* Ps  = (const float4*)(P  + (size_t)s * 512);
  const float4* Hd  = (const float4*)(H  + (size_t)d * HF);
  const float4* H2d = (const float4*)(H2 + (size_t)d * HF);
  float acc = 0.f;
  for (int q = 0; q < 64; q++) { float4 p = Ps[q];      float4 h = Hd[q];
    acc += p.x*h.x + p.y*h.y + p.z*h.z + p.w*h.w; }
  for (int q = 0; q < 64; q++) { float4 p = Ps[64 + q]; float4 h = H2d[q];
    acc += p.x*h.x + p.y*h.y + p.z*h.z + p.w*h.w; }
  float logit = acc + bb[0];
  out[1 + m] = logit;
  out[1 + ME + m] = (logit >= 0.f) ? 1.f : 0.f;
  float y = labels[m];
  float lt = fmaxf(logit, 0.f) - logit * y + log1pf(__expf(-fabsf(logit)));
  red[t] = lt; __syncthreads();
  for (int ss = 128; ss > 0; ss >>= 1) { if (t < ss) red[t] += red[t + ss]; __syncthreads(); }
  if (t == 0) lossp[blockIdx.x] = red[0];
}
__global__ void loss_fin(const float* __restrict__ lossp, float* __restrict__ out) {
  __shared__ float red[256];
  int t = threadIdx.x;
  red[t] = lossp[t]; __syncthreads();
  for (int s = 128; s > 0; s >>= 1) { if (t < s) red[t] += red[t + s]; __syncthreads(); }
  if (t == 0) out[0] = red[0] / (float)ME;
}

extern "C" void kernel_launch(void* const* d_in, const int* in_sizes, int n_in,
                              void* d_out, int out_size, void* d_ws, size_t ws_size,
                              hipStream_t stream) {
  (void)in_sizes; (void)n_in; (void)out_size; (void)ws_size;
  const int*   src    = (const int*)  d_in[0];
  const int*   dst    = (const int*)  d_in[1];
  const float* labels = (const float*)d_in[2];
  const float* adj    = (const float*)d_in[3];
  const float* x      = (const float*)d_in[4];
  const float* W1_0   = (const float*)d_in[5];
  const float* W1_1   = (const float*)d_in[6];
  const float* W2_0   = (const float*)d_in[7];
  const float* W2_1   = (const float*)d_in[8];
  const float* Wm     = (const float*)d_in[9];
  const float* a2     = (const float*)d_in[11];
  const float* bilw   = (const float*)d_in[12];
  const float* bilb   = (const float*)d_in[13];
  float* out = (float*)d_out;

  float* ws      = (float*)d_ws;
  float* deg1    = ws;                 // 8192
  float* s2      = ws + 8192;          // 8192
  float* w2      = ws + 16384;         // 512
  float* TS      = ws + 16896;         // 256
  float* SH      = ws + 17152;         // 256
  float* cpart   = ws + 17408;         // 64*256
  float* lossp   = ws + 33792;         // 256
  int*   hubs    = (int*)(ws + 34048); // 16
  int*   hubflag = (int*)(ws + 34064); // 8192
  float* degpart = ws + 43008;         // 32*8192
  float* Yb  = ws + 524288;            // 8192*256
  float* Ab  = Yb  + 2097152;          // 8192*256 (Z1 / Z2)
  float* Hb  = Ab  + 2097152;          // 8192*256 (h)
  float* H2b = Hb  + 2097152;          // 8192*256 (h_)
  float* Pb  = H2b + 2097152;          // 8192*512 (x @ bil_w)

  // deg for _adj_norm(adj)
  adj_colsum_part<<<dim3(32, 32), 256, 0, stream>>>(adj, degpart);
  adj_deg_fin<<<32, 256, 0, stream>>>(degpart, deg1);

  // GCN1: z1 = elu((adj@Y + Y)/deg), Y = x@W1_0 ; then layer 2 -> h
  gemm_bf16<<<dim3(1, 256), 256, 0, stream>>>(x,   512,  W1_0, 256, Yb, 256, 512,  0, nullptr, nullptr);
  gemm_bf16<<<dim3(1, 256), 256, 0, stream>>>(adj, 8192, Yb,   256, Ab, 256, 8192, 1, Yb, deg1);
  gemm_bf16<<<dim3(1, 256), 256, 0, stream>>>(Ab,  256,  W1_1, 256, Yb, 256, 256,  0, nullptr, nullptr);
  gemm_bf16<<<dim3(1, 256), 256, 0, stream>>>(adj, 8192, Yb,   256, Hb, 256, 8192, 1, Yb, deg1);

  // similarity graph structure: top-16 of s2 = x @ (W@a2)
  w_dot_a2<<<2, 256, 0, stream>>>(Wm, a2, w2);
  s2_kernel<<<NND, 256, 0, stream>>>(x, w2, s2);
  topk_kernel<<<1, 256, 0, stream>>>(s2, hubs, hubflag);

  // GCN2 via hub structure
  gemm_bf16<<<dim3(1, 256), 256, 0, stream>>>(x, 512, W2_0, 256, Yb, 256, 512, 0, nullptr, nullptr);
  colsum_part<<<64, 256, 0, stream>>>(Yb, cpart);
  colsum_fin<<<1, 256, 0, stream>>>(cpart, Yb, hubs, TS, SH);
  gcn2_ew<<<NND, 256, 0, stream>>>(Yb, TS, SH, hubflag, Ab);
  gemm_bf16<<<dim3(1, 256), 256, 0, stream>>>(Ab, 256, W2_1, 256, Yb, 256, 256, 0, nullptr, nullptr);
  colsum_part<<<64, 256, 0, stream>>>(Yb, cpart);
  colsum_fin<<<1, 256, 0, stream>>>(cpart, Yb, hubs, TS, SH);
  gcn2_ew<<<NND, 256, 0, stream>>>(Yb, TS, SH, hubflag, H2b);

  // bilinear head: P = x @ bil_w, then per-edge dot with [h, h_]
  gemm_bf16<<<dim3(2, 256), 256, 0, stream>>>(x, 512, bilw, 512, Pb, 512, 512, 0, nullptr, nullptr);
  edge_kernel<<<256, 256, 0, stream>>>(src, dst, labels, Pb, Hb, H2b, bilb, out, lossp);
  loss_fin<<<1, 256, 0, stream>>>(lossp, out);
}
